// Net_28028956573872
// MI455X (gfx1250) — matrix operand order repaired and checked by reading, not compiled, not run on hardware
//
#include <hip/hip_runtime.h>
#include <hip/hip_bf16.h>
#include <math.h>

#define BB 4
#define JJ 2048
#define CC 128
#define KDIM 32
#define JP 2049            // padded logical dim (slack row/col)
#define LDP 2064           // row stride in floats (multiple of 16 -> aligned)
#define NIT 20
#define ALPHA_OFF 0.5f

typedef __attribute__((ext_vector_type(16))) _Float16 v16h;
typedef __attribute__((ext_vector_type(8)))  float    v8f;

struct alignas(16) H8 { _Float16 h[8]; };

static __device__ __forceinline__ v8f wmma16(v16h a, v16h b, v8f c) {
  return __builtin_amdgcn_wmma_f32_16x16x32_f16(false, a, false, b, (short)0, c,
                                                false, false);
}

// A-fragment: 16x32 f16, row m = m0+(lane&15); lane<16 holds K={0..7,16..23},
// lane>=16 holds K={8..15,24..31} (ISA 7.12.2).
static __device__ __forceinline__ v16h load_fragA_h(const _Float16* base, int ld,
                                                    int m0, int k0, int lane) {
  int hl = lane >> 4;
  const _Float16* p = base + (size_t)(m0 + (lane & 15)) * ld + k0 + hl * 8;
  H8 c0 = *(const H8*)(p);
  H8 c1 = *(const H8*)(p + 16);
  v16h v;
#pragma unroll
  for (int i = 0; i < 8; ++i) { v[i] = c0.h[i]; v[i + 8] = c1.h[i]; }
  return v;
}

// B-fragment from transposed storage Bt[n][k] (K contiguous per column):
// col n = n0+(lane&15); lane<16 holds K={0..7,16..23}, lane>=16 holds K={8..15,24..31}.
static __device__ __forceinline__ v16h load_fragB_h(const _Float16* baseT, int ld,
                                                    int n0, int k0, int lane) {
  int hl = lane >> 4;
  const _Float16* p = baseT + (size_t)(n0 + (lane & 15)) * ld + k0 + hl * 8;
  H8 c0 = *(const H8*)(p);
  H8 c1 = *(const H8*)(p + 16);
  v16h v;
#pragma unroll
  for (int i = 0; i < 8; ++i) { v[i] = c0.h[i]; v[i + 8] = c1.h[i]; }
  return v;
}

// A-fragment converted on the fly from f32 rows (row stride must be 16B-aligned).
static __device__ __forceinline__ v16h load_fragA_f32(const float* base, int ld,
                                                      int m0, int k0, int lane) {
  int hl = lane >> 4;
  const float* p = base + (size_t)(m0 + (lane & 15)) * ld + k0 + hl * 8;
  float4 a0 = *(const float4*)(p);
  float4 a1 = *(const float4*)(p + 4);
  float4 b0 = *(const float4*)(p + 16);
  float4 b1 = *(const float4*)(p + 20);
  v16h v;
  v[0]=(_Float16)a0.x; v[1]=(_Float16)a0.y; v[2]=(_Float16)a0.z; v[3]=(_Float16)a0.w;
  v[4]=(_Float16)a1.x; v[5]=(_Float16)a1.y; v[6]=(_Float16)a1.z; v[7]=(_Float16)a1.w;
  v[8]=(_Float16)b0.x; v[9]=(_Float16)b0.y; v[10]=(_Float16)b0.z; v[11]=(_Float16)b0.w;
  v[12]=(_Float16)b1.x; v[13]=(_Float16)b1.y; v[14]=(_Float16)b1.z; v[15]=(_Float16)b1.w;
  return v;
}

static __device__ __forceinline__ void lse_combine(float& m, float& s,
                                                   float m2, float s2) {
  if (m2 > m) { s = s * __expf(m - m2) + s2; m = m2; }
  else        { s = s + s2 * __expf(m2 - m); }
}

// ---------------------------------------------------------------- y_avg
__global__ void yavg_kernel(const float* __restrict__ ref, float* __restrict__ yavg) {
  int b = blockIdx.x, c = threadIdx.x;
  float s = 0.f;
  const float* p = ref + (size_t)b * JJ * CC + c;
  for (int j = 0; j < JJ; ++j) s += p[(size_t)j * CC];
  yavg[b * CC + c] = s * (1.0f / JJ);
}

// ----------------------------------------- transposed f16 weights + q/k bias
__global__ void prepw_kernel(const float* __restrict__ Wq, const float* __restrict__ bq,
                             const float* __restrict__ Wk, const float* __restrict__ bk,
                             const float* __restrict__ yavg,
                             _Float16* __restrict__ wqt, _Float16* __restrict__ wkt,
                             float* __restrict__ qb, float* __restrict__ kb) {
  int tid = threadIdx.x;
  for (int idx = tid; idx < KDIM * CC; idx += 256) {
    int n = idx / CC, c = idx % CC;              // wqt[n][c] = Wq[c][n]
    wqt[idx] = (_Float16)Wq[c * KDIM + n];
    wkt[idx] = (_Float16)Wk[c * KDIM + n];
  }
  for (int idx = tid; idx < BB * KDIM; idx += 256) {
    int b = idx / KDIM, n = idx % KDIM;
    float sq = bq[n], sk = bk[n];
    for (int c = 0; c < CC; ++c) {
      float y = yavg[b * CC + c];
      sq += y * Wq[(CC + c) * KDIM + n];
      sk += y * Wk[(CC + c) * KDIM + n];
    }
    qb[idx] = sq; kb[idx] = sk;
  }
}

// ------------------------------------------------- f32 -> f16 conversions
__global__ void convert_kernel(const float* __restrict__ src, const float* __restrict__ ref,
                               _Float16* __restrict__ src16, _Float16* __restrict__ ref16,
                               _Float16* __restrict__ srcT16) {
  size_t idx = (size_t)blockIdx.x * 256 + threadIdx.x;   // < BB*JJ*CC
  int b = (int)(idx / ((size_t)JJ * CC));
  int r = (int)(idx % ((size_t)JJ * CC));
  int j = r / CC, c = r % CC;
  float s = src[idx];
  src16[idx] = (_Float16)s;
  srcT16[((size_t)b * CC + c) * JJ + j] = (_Float16)s;
  ref16[idx] = (_Float16)ref[idx];
}

// ------------------------------------------------- squared row norms (f16 in)
__global__ void norm16_kernel(const _Float16* __restrict__ x, float* __restrict__ out) {
  __shared__ float sm[128];
  int row = blockIdx.x, tid = threadIdx.x;
  float v = (float)x[(size_t)row * CC + tid];
  sm[tid] = v * v;
  __syncthreads();
  for (int off = 64; off > 0; off >>= 1) {
    if (tid < off) sm[tid] += sm[tid + off];
    __syncthreads();
  }
  if (tid == 0) out[row] = sm[0];
}

// ------------------------------------------- q/k projection GEMM (WMMA, K=128)
__global__ void qk_kernel(const _Float16* __restrict__ src16,
                          const _Float16* __restrict__ wqt, const _Float16* __restrict__ wkt,
                          const float* __restrict__ qb, const float* __restrict__ kb,
                          _Float16* __restrict__ q16, _Float16* __restrict__ k16) {
  int tid = threadIdx.x, lane = tid & 31, warp = tid >> 5;
  int wid = blockIdx.x * 8 + warp;          // 0..511
  int b = wid >> 7, mt = wid & 127, m0 = mt * 16;
  int hl = lane >> 4, ln = lane & 15;
  const _Float16* A = src16 + (size_t)b * JJ * CC;
  v8f aq[2], ak[2];
#pragma unroll
  for (int t = 0; t < 2; ++t) {
    float qv = qb[b * KDIM + t * 16 + ln];
    float kv = kb[b * KDIM + t * 16 + ln];
#pragma unroll
    for (int r = 0; r < 8; ++r) { aq[t][r] = qv; ak[t][r] = kv; }
  }
#pragma unroll
  for (int kk = 0; kk < 4; ++kk) {
    v16h a = load_fragA_h(A, CC, m0, kk * 32, lane);
#pragma unroll
    for (int t = 0; t < 2; ++t) {
      v16h bqf = load_fragB_h(wqt, CC, t * 16, kk * 32, lane);
      aq[t] = wmma16(a, bqf, aq[t]);
      v16h bkf = load_fragB_h(wkt, CC, t * 16, kk * 32, lane);
      ak[t] = wmma16(a, bkf, ak[t]);
    }
  }
#pragma unroll
  for (int t = 0; t < 2; ++t)
#pragma unroll
    for (int r = 0; r < 8; ++r) {
      int M = m0 + r + hl * 8, N = t * 16 + ln;
      q16[((size_t)b * JJ + M) * KDIM + N] = (_Float16)aq[t][r];
      k16[((size_t)b * JJ + M) * KDIM + N] = (_Float16)ak[t][r];
    }
}

// ----------------------------------- attention scores q @ k^T (WMMA, K=32)
__global__ void scores_kernel(const _Float16* __restrict__ q16,
                              const _Float16* __restrict__ k16,
                              float* __restrict__ la) {
  int tid = threadIdx.x, lane = tid & 31, warp = tid >> 5;
  int wid = blockIdx.x * 8 + warp;          // 0..16383
  int b = wid >> 12, rem = wid & 4095, mt = rem >> 5, nt = rem & 31;
  int m0 = mt * 16, n0 = nt * 64;
  int hl = lane >> 4, ln = lane & 15;
  const _Float16* Q = q16 + (size_t)b * JJ * KDIM;
  const _Float16* K = k16 + (size_t)b * JJ * KDIM;
  float* lab = la + (size_t)b * JP * LDP;
  v16h a = load_fragA_h(Q, KDIM, m0, 0, lane);
#pragma unroll
  for (int t = 0; t < 4; ++t) {
    v16h bf = load_fragB_h(K, KDIM, n0 + t * 16, 0, lane);
    v8f acc = {};
    acc = wmma16(a, bf, acc);
#pragma unroll
    for (int r = 0; r < 8; ++r) {
      int M = m0 + r + hl * 8;
      lab[(size_t)M * LDP + (n0 + t * 16 + ln)] = acc[r];
    }
  }
}

// ------------------------------------------------ row softmax (in place)
__global__ void softmax_kernel(float* __restrict__ la) {
  __shared__ float red[256];
  __shared__ float sM, sS;
  int row = blockIdx.x, tid = threadIdx.x;
  int b = row / JJ, j = row % JJ;
  float* p = la + (size_t)b * JP * LDP + (size_t)j * LDP;
  float mx = -INFINITY;
  for (int c = tid; c < JJ; c += 256) mx = fmaxf(mx, p[c]);
  red[tid] = mx; __syncthreads();
  for (int off = 128; off > 0; off >>= 1) {
    if (tid < off) red[tid] = fmaxf(red[tid], red[tid + off]);
    __syncthreads();
  }
  if (tid == 0) sM = red[0];
  __syncthreads();
  float s = 0.f;
  for (int c = tid; c < JJ; c += 256) s += __expf(p[c] - sM);
  red[tid] = s; __syncthreads();
  for (int off = 128; off > 0; off >>= 1) {
    if (tid < off) red[tid] += red[tid + off];
    __syncthreads();
  }
  if (tid == 0) sS = 1.0f / red[0];
  __syncthreads();
  for (int c = tid; c < JJ; c += 256) p[c] = __expf(p[c] - sM) * sS;
}

// -------------------------------- feat_w = att @ feat_src (WMMA, K=2048)
__global__ void fw_kernel(const float* __restrict__ la,
                          const _Float16* __restrict__ srcT16,
                          _Float16* __restrict__ w16) {
  int tid = threadIdx.x, lane = tid & 31, warp = tid >> 5;
  int wid = blockIdx.x * 8 + warp;          // 0..1023
  int b = wid >> 8, rem = wid & 255, mt = rem >> 1, ng = rem & 1;
  int m0 = mt * 16, n0 = ng * 64;
  int hl = lane >> 4, ln = lane & 15;
  const float* lab = la + (size_t)b * JP * LDP;
  const _Float16* Bt = srcT16 + (size_t)b * CC * JJ;
  v8f acc[4] = {};
  for (int ks = 0; ks < 64; ++ks) {
    int k0 = ks * 32;
    __builtin_prefetch(lab + (size_t)(m0 + (lane & 15)) * LDP + k0 + 64, 0, 0);
    v16h a = load_fragA_f32(lab, LDP, m0, k0, lane);
#pragma unroll
    for (int t = 0; t < 4; ++t) {
      v16h bf = load_fragB_h(Bt, JJ, n0 + t * 16, k0, lane);
      acc[t] = wmma16(a, bf, acc[t]);
    }
  }
#pragma unroll
  for (int t = 0; t < 4; ++t)
#pragma unroll
    for (int r = 0; r < 8; ++r) {
      int M = m0 + r + hl * 8, N = n0 + t * 16 + ln;
      w16[((size_t)b * JJ + M) * CC + N] = (_Float16)acc[t][r];
    }
}

// ----------------- log_alpha = 2*w.ref - |w|^2 - |ref|^2 + alpha (WMMA, K=128)
__global__ void dist_kernel(const _Float16* __restrict__ w16,
                            const _Float16* __restrict__ ref16,
                            const float* __restrict__ wn, const float* __restrict__ rn,
                            float* __restrict__ la) {
  int tid = threadIdx.x, lane = tid & 31, warp = tid >> 5;
  int wid = blockIdx.x * 8 + warp;          // 0..16383
  int b = wid >> 12, rem = wid & 4095, mt = rem >> 5, nt = rem & 31;
  int m0 = mt * 16, n0 = nt * 64;
  int hl = lane >> 4, ln = lane & 15;
  const _Float16* A = w16 + (size_t)b * JJ * CC;
  const _Float16* R = ref16 + (size_t)b * JJ * CC;
  float* lab = la + (size_t)b * JP * LDP;
  v8f acc[4] = {};
#pragma unroll
  for (int kk = 0; kk < 4; ++kk) {
    v16h a = load_fragA_h(A, CC, m0, kk * 32, lane);
#pragma unroll
    for (int t = 0; t < 4; ++t) {
      v16h bf = load_fragB_h(R, CC, n0 + t * 16, kk * 32, lane);
      acc[t] = wmma16(a, bf, acc[t]);
    }
  }
#pragma unroll
  for (int t = 0; t < 4; ++t)
#pragma unroll
    for (int r = 0; r < 8; ++r) {
      int M = m0 + r + hl * 8, N = n0 + t * 16 + ln;
      float v = 2.0f * acc[t][r] - wn[b * JJ + M] - rn[b * JJ + N] + ALPHA_OFF;
      lab[(size_t)M * LDP + N] = v;
    }
}

// ------------------------------------------------ slack row/col = 0
__global__ void slack_kernel(float* __restrict__ la) {
  int idx = blockIdx.x * 256 + threadIdx.x;
  if (idx >= BB * JP) return;
  int b = idx / JP, p = idx % JP;
  float* lab = la + (size_t)b * JP * LDP;
  lab[(size_t)JJ * LDP + p] = 0.0f;      // slack row
  lab[(size_t)p * LDP + JJ] = 0.0f;      // slack col
}

// --------------------- Sinkhorn row pass: rows 0..J-1 over 2049 columns
__global__ void rownorm_kernel(float* __restrict__ la) {
  __shared__ float rm[256], rs[256];
  __shared__ float lse;
  int row = blockIdx.x, tid = threadIdx.x;
  int b = row / JJ, j = row % JJ;
  float* p = la + (size_t)b * JP * LDP + (size_t)j * LDP;
  float m = -INFINITY, s = 0.f;
  for (int c = tid; c < JP; c += 256) {
    float x = p[c];
    if (x > m) { s = s * __expf(m - x) + 1.f; m = x; }
    else       { s += __expf(x - m); }
  }
  rm[tid] = m; rs[tid] = s; __syncthreads();
  for (int off = 128; off > 0; off >>= 1) {
    if (tid < off) lse_combine(rm[tid], rs[tid], rm[tid + off], rs[tid + off]);
    __syncthreads();
  }
  if (tid == 0) lse = rm[0] + __logf(rs[0]);
  __syncthreads();
  for (int c = tid; c < JP; c += 256) p[c] -= lse;
}

// --------------------- Sinkhorn col pass: cols 0..J-1 over 2049 rows
__global__ void colnorm_kernel(float* __restrict__ la) {
  __shared__ float cm[256], cs[256];
  __shared__ float clse[32];
  int tid = threadIdx.x, tx = tid & 31, ty = tid >> 5;
  int b = blockIdx.x >> 6, cg = blockIdx.x & 63;
  int col = cg * 32 + tx;
  float* lab = la + (size_t)b * JP * LDP;
  float m = -INFINITY, s = 0.f;
  for (int r = ty; r < JP; r += 8) {
    float x = lab[(size_t)r * LDP + col];
    if (x > m) { s = s * __expf(m - x) + 1.f; m = x; }
    else       { s += __expf(x - m); }
  }
  cm[ty * 32 + tx] = m; cs[ty * 32 + tx] = s;
  __syncthreads();
  if (ty == 0) {
    float mm = cm[tx], ss = cs[tx];
#pragma unroll
    for (int g = 1; g < 8; ++g) lse_combine(mm, ss, cm[g * 32 + tx], cs[g * 32 + tx]);
    clse[tx] = mm + __logf(ss);
  }
  __syncthreads();
  float l = clse[tx];
  for (int r = ty; r < JP; r += 8) lab[(size_t)r * LDP + col] -= l;
}

// ------------------------------------------------ final exp, strip slack
__global__ void exp_kernel(const float* __restrict__ la, float* __restrict__ out) {
  size_t idx = (size_t)blockIdx.x * 256 + threadIdx.x;   // < BB*JJ*JJ
  int b = (int)(idx / ((size_t)JJ * JJ));
  size_t r = idx % ((size_t)JJ * JJ);
  int j = (int)(r / JJ), k = (int)(r % JJ);
  out[idx] = __expf(la[(size_t)b * JP * LDP + (size_t)j * LDP + k]);
}

extern "C" void kernel_launch(void* const* d_in, const int* in_sizes, int n_in,
                              void* d_out, int out_size, void* d_ws, size_t ws_size,
                              hipStream_t stream) {
  (void)in_sizes; (void)n_in; (void)out_size; (void)ws_size;
  const float* feat_src = (const float*)d_in[0];
  const float* feat_ref = (const float*)d_in[1];
  const float* Wq = (const float*)d_in[2];
  const float* bq = (const float*)d_in[3];
  const float* Wk = (const float*)d_in[4];
  const float* bk = (const float*)d_in[5];
  float* out = (float*)d_out;

  char* w = (char*)d_ws;
  size_t off = 0;
  auto take = [&](size_t bytes) { char* p = w + off; off += (bytes + 255) & ~(size_t)255; return p; };
  float*    la     = (float*)   take((size_t)BB * JP * LDP * 4);
  _Float16* src16  = (_Float16*)take((size_t)BB * JJ * CC * 2);
  _Float16* ref16  = (_Float16*)take((size_t)BB * JJ * CC * 2);
  _Float16* srcT16 = (_Float16*)take((size_t)BB * CC * JJ * 2);
  _Float16* w16    = (_Float16*)take((size_t)BB * JJ * CC * 2);
  _Float16* q16    = (_Float16*)take((size_t)BB * JJ * KDIM * 2);
  _Float16* k16    = (_Float16*)take((size_t)BB * JJ * KDIM * 2);
  _Float16* wqt    = (_Float16*)take((size_t)KDIM * CC * 2);
  _Float16* wkt    = (_Float16*)take((size_t)KDIM * CC * 2);
  float*    qb     = (float*)   take((size_t)BB * KDIM * 4);
  float*    kb     = (float*)   take((size_t)BB * KDIM * 4);
  float*    yavg   = (float*)   take((size_t)BB * CC * 4);
  float*    rnorm  = (float*)   take((size_t)BB * JJ * 4);
  float*    wnorm  = (float*)   take((size_t)BB * JJ * 4);

  yavg_kernel<<<BB, 128, 0, stream>>>(feat_ref, yavg);
  prepw_kernel<<<1, 256, 0, stream>>>(Wq, bq, Wk, bk, yavg, wqt, wkt, qb, kb);
  convert_kernel<<<(BB * JJ * CC) / 256, 256, 0, stream>>>(feat_src, feat_ref,
                                                           src16, ref16, srcT16);
  norm16_kernel<<<BB * JJ, 128, 0, stream>>>(ref16, rnorm);
  qk_kernel<<<64, 256, 0, stream>>>(src16, wqt, wkt, qb, kb, q16, k16);
  scores_kernel<<<2048, 256, 0, stream>>>(q16, k16, la);
  softmax_kernel<<<BB * JJ, 256, 0, stream>>>(la);
  fw_kernel<<<128, 256, 0, stream>>>(la, srcT16, w16);
  norm16_kernel<<<BB * JJ, 128, 0, stream>>>(w16, wnorm);
  dist_kernel<<<2048, 256, 0, stream>>>(w16, ref16, wnorm, rnorm, la);
  slack_kernel<<<(BB * JP + 255) / 256, 256, 0, stream>>>(la);
  for (int it = 0; it < NIT; ++it) {
    rownorm_kernel<<<BB * JJ, 256, 0, stream>>>(la);
    colnorm_kernel<<<BB * 64, 256, 0, stream>>>(la);
  }
  exp_kernel<<<(BB * JJ * JJ) / 256, 256, 0, stream>>>(la, out);
}
